// VisionAttention_37168646979912
// MI455X (gfx1250) — matrix operand order repaired and checked by reading, not compiled, not run on hardware
//
#include <hip/hip_runtime.h>
#include <hip/hip_bf16.h>

// ---------------------------------------------------------------------------
// VisionAttention for MI455X (gfx1250, wave32, WMMA + async/TDM staging).
//   T=4096 tokens, D=1536 hidden, H=12 heads, HD=128 head dim, 4 segments.
// Pipeline: f32->f16 convert, WMMA QKV GEMM (async-to-LDS staging), in-place
// RoPE (+q scaling), flash attention (TDM tensor_load_to_lds for K chunks),
// WMMA out-projection to f32.
// ---------------------------------------------------------------------------

#define T_TOK  4096
#define DMODEL 1536
#define NH     12
#define HDIM   128
#define QKV_LD (3 * DMODEL)   // 4608

typedef __attribute__((ext_vector_type(16))) _Float16 v16h;
typedef __attribute__((ext_vector_type(8)))  float    v8f;
typedef __attribute__((ext_vector_type(4)))  unsigned u32x4;
typedef __attribute__((ext_vector_type(8)))  unsigned u32x8;
typedef __attribute__((ext_vector_type(4)))  int      v4i;

typedef __attribute__((address_space(1))) v4i* gptr_v4i;   // global int4*
typedef __attribute__((address_space(3))) v4i* lptr_v4i;   // LDS int4*

union F16x16 { v16h v; uint4 u[2]; };
union H8     { uint4 u; _Float16 h[8]; };

// ---------------------------------------------------------------------------
// CDNA5 async global->LDS 16B copy (ASYNCcnt path), with safe fallback.
// Probe-verified: builtin takes (int4 addrspace(1)*, int4 addrspace(3)*, imm, imm).
// ---------------------------------------------------------------------------
__device__ __forceinline__ void ld_async_b128(const _Float16* g, _Float16* l) {
#if __has_builtin(__builtin_amdgcn_global_load_async_to_lds_b128)
  __builtin_amdgcn_global_load_async_to_lds_b128(
      (gptr_v4i)(void*)g, (lptr_v4i)(void*)l, 0, 0);
#else
  *(uint4*)l = *(const uint4*)g;
#endif
}
__device__ __forceinline__ void wait_async0() {
#if __has_builtin(__builtin_amdgcn_global_load_async_to_lds_b128)
#if __has_builtin(__builtin_amdgcn_s_wait_asynccnt)
  __builtin_amdgcn_s_wait_asynccnt(0);
#else
  asm volatile("s_wait_asynccnt 0x0" ::: "memory");
#endif
#endif
}

// ---------------------------------------------------------------------------
// fp32 -> f16 conversion
// ---------------------------------------------------------------------------
__global__ void cvt_f32_to_f16(const float* __restrict__ in,
                               _Float16* __restrict__ out, int n) {
  int i = blockIdx.x * blockDim.x + threadIdx.x;
  int stride = gridDim.x * blockDim.x;
  for (; i < n; i += stride) out[i] = (_Float16)in[i];
}

// ---------------------------------------------------------------------------
// C(MxN) = A(MxK,f16) * B(NxK,f16)^T  via v_wmma_f32_16x16x32_f16.
// Block tile 128x128, BK=32, 256 threads = 8 waves arranged 4(M) x 2(N):
// wave owns 32(M) x 64(N) -> 2x4 v8f accumulators, 8 WMMA per K-step.
// A-frag: lane holds row m=lane%16; halves 0-7 = K(8*hi..), 8-15 = K(16+8*hi..)
// B-frag: lane holds col n=lane%16; halves 0-7 = K(8*hi..), 8-15 = K(16+8*hi..)
// ---------------------------------------------------------------------------
template <bool F16OUT>
__global__ __launch_bounds__(256)
void gemm_wmma(const _Float16* __restrict__ A, const _Float16* __restrict__ B,
               _Float16* __restrict__ Ch, float* __restrict__ Cf,
               int M, int N, int K) {
  __shared__ _Float16 As[128][40];   // rows padded to 40 halves (80B, 16B-aligned)
  __shared__ _Float16 Bs[128][40];
  const int tid  = threadIdx.x;
  const int wave = tid >> 5;
  const int lane = tid & 31;
  const int lid  = lane & 15;
  const int hi   = lane >> 4;
  const int wm   = (wave >> 1) * 32;   // 0,32,64,96
  const int wn   = (wave & 1) * 64;    // 0,64
  const int m0 = blockIdx.x * 128;
  const int n0 = blockIdx.y * 128;

  v8f acc[2][4] = {};

  for (int k0 = 0; k0 < K; k0 += 32) {
    // stage 128x32 A and B tiles: 512 x 16B units each, async-to-LDS
    for (int u = tid; u < 512; u += 256) {
      int row = u >> 2, c8 = (u & 3) * 8;
      ld_async_b128(A + (size_t)(m0 + row) * K + k0 + c8, &As[row][c8]);
      ld_async_b128(B + (size_t)(n0 + row) * K + k0 + c8, &Bs[row][c8]);
    }
    wait_async0();
    __syncthreads();

    F16x16 fa[2];
#pragma unroll
    for (int mt = 0; mt < 2; ++mt) {
      fa[mt].u[0] = *(const uint4*)&As[wm + 16 * mt + lid][8 * hi];
      fa[mt].u[1] = *(const uint4*)&As[wm + 16 * mt + lid][16 + 8 * hi];
    }
#pragma unroll
    for (int nt = 0; nt < 4; ++nt) {
      F16x16 fb;
      fb.u[0] = *(const uint4*)&Bs[wn + 16 * nt + lid][8 * hi];
      fb.u[1] = *(const uint4*)&Bs[wn + 16 * nt + lid][16 + 8 * hi];
#pragma unroll
      for (int mt = 0; mt < 2; ++mt)
        acc[mt][nt] = __builtin_amdgcn_wmma_f32_16x16x32_f16(
            false, fa[mt].v, false, fb.v, (short)0, acc[mt][nt], false, false);
    }
    __syncthreads();
  }

  // C layout: VGPR j of v8f holds (m = j + 8*hi, n = lane%16)
#pragma unroll
  for (int mt = 0; mt < 2; ++mt)
#pragma unroll
    for (int nt = 0; nt < 4; ++nt)
#pragma unroll
      for (int j = 0; j < 8; ++j) {
        int r = m0 + wm + 16 * mt + j + 8 * hi;
        int c = n0 + wn + 16 * nt + lid;
        if (F16OUT) Ch[(size_t)r * N + c] = (_Float16)acc[mt][nt][j];
        else        Cf[(size_t)r * N + c] = acc[mt][nt][j];
      }
}

// ---------------------------------------------------------------------------
// Vision RoPE applied in place on f16 qkv buffer; folds HD^-0.5 into Q.
// ---------------------------------------------------------------------------
__global__ void rope_kernel(_Float16* __restrict__ qkv,
                            const float* __restrict__ rpe) {
  int idx = blockIdx.x * blockDim.x + threadIdx.x;
  if (idx >= T_TOK * NH * 64) return;
  int d = idx & 63;
  int h = (idx >> 6) % NH;
  int t = idx / (NH * 64);
  float ang = rpe[t * 64 + d];
  float c = cosf(ang), s = sinf(ang);
  const float scale = 0.08838834764831845f;  // 128^-0.5 folded into Q
  _Float16* row = qkv + (size_t)t * QKV_LD;
  {
    _Float16* q = row + h * HDIM;
    float x1 = (float)q[d], x2 = (float)q[d + 64];
    q[d]      = (_Float16)((x1 * c - x2 * s) * scale);
    q[d + 64] = (_Float16)((x2 * c + x1 * s) * scale);
  }
  {
    _Float16* k = row + DMODEL + h * HDIM;
    float x1 = (float)k[d], x2 = (float)k[d + 64];
    k[d]      = (_Float16)(x1 * c - x2 * s);
    k[d + 64] = (_Float16)(x2 * c + x1 * s);
  }
}

// ---------------------------------------------------------------------------
// Flash attention with TDM K-chunk staging.
// Block = 128 threads (4 waves) = 64 query rows of one head; keys in 32-chunks.
// K chunk (32x128 f16) is DMA'd into LDS by wave 0 via tensor_load_to_lds with
// a 2D D# (data_size=2B, tile 128x32, tensor row stride 4608), completion via
// s_wait_tensorcnt. V chunk is manually transposed into LDS for the B-operand.
// ---------------------------------------------------------------------------
__global__ __launch_bounds__(128)
void attn_kernel(const _Float16* __restrict__ qkv, const int* __restrict__ cu,
                 _Float16* __restrict__ out) {
  __shared__ _Float16 Ks[32][128];      // K chunk, [key][d] (TDM destination)
  __shared__ _Float16 Vts[HDIM][40];    // V chunk transposed, [d][key] (pad 8)
  __shared__ _Float16 Ps[4][16][32];    // per-wave P redistribute tile
  const int h    = blockIdx.y;
  const int tid  = threadIdx.x;
  const int wave = tid >> 5;
  const int lane = tid & 31;
  const int lid  = lane & 15;
  const int hi   = lane >> 4;
  const int qt0  = blockIdx.x * 64 + wave * 16;

  // segment bounds for this query tile (boundaries are multiples of 64 here)
  int lo = 0, seg_end = T_TOK;
  for (int s = 0; s < 4; ++s)
    if (qt0 >= cu[s] && qt0 < cu[s + 1]) { lo = cu[s]; seg_end = cu[s + 1]; }

  // Q fragments (A layout), read from qkv (RoPE + scale already applied)
  F16x16 qf[4];
  {
    const _Float16* qrow = qkv + (size_t)(qt0 + lid) * QKV_LD + h * HDIM;
#pragma unroll
    for (int dc = 0; dc < 4; ++dc) {
      qf[dc].u[0] = *(const uint4*)(qrow + dc * 32 + 8 * hi);
      qf[dc].u[1] = *(const uint4*)(qrow + dc * 32 + 16 + 8 * hi);
    }
  }

  // TDM descriptor pieces that are loop-invariant
  const unsigned lds_ks =
      (unsigned)(uintptr_t)(__attribute__((address_space(3))) void*)&Ks[0][0];

  float m_run[8], l_run[8];
  v8f O[8] = {};
#pragma unroll
  for (int j = 0; j < 8; ++j) { m_run[j] = -1e30f; l_run[j] = 0.f; }

  for (int kk = lo; kk < seg_end; kk += 32) {
    // ---- K chunk via Tensor Data Mover (wave 0 issues; TENSORcnt per-wave)
    if (wave == 0) {
      unsigned long long ga = (unsigned long long)(uintptr_t)(
          qkv + (size_t)kk * QKV_LD + DMODEL + (size_t)h * HDIM);
      u32x4 g0;
      g0[0] = 1u;                                   // count=1, user mode
      g0[1] = lds_ks;                               // lds_addr
      g0[2] = (unsigned)(ga & 0xffffffffu);         // global_addr[31:0]
      g0[3] = (unsigned)((ga >> 32) & 0x01ffffffu)  // global_addr[56:32]
              | (2u << 30);                         // type=2 ("image")
      u32x8 g1;
      g1[0] = 1u << 16;                             // data_size=2B, no multicast
      g1[1] = ((unsigned)QKV_LD & 0xffffu) << 16;   // tensor_dim0 lo16
      g1[2] = ((unsigned)QKV_LD >> 16) | ((unsigned)T_TOK << 16);  // dim0 hi/dim1 lo
      g1[3] = ((unsigned)T_TOK >> 16) | (128u << 16);              // dim1 hi, tile_dim0=128
      g1[4] = 32u;                                  // tile_dim1=32, tile_dim2=0
      g1[5] = (unsigned)QKV_LD;                     // tensor_dim0_stride lo32
      g1[6] = 0u;                                   // stride0 hi, stride1 lo
      g1[7] = 0u;                                   // stride1 hi
      u32x4 gz = {0u, 0u, 0u, 0u};                  // groups 2/3 unused (2D)
      asm volatile("tensor_load_to_lds %0, %1, %2, %3"
                   :: "s"(g0), "s"(g1), "s"(gz), "s"(gz) : "memory");
    }

    // ---- V chunk: manual transposed staging [d][key], 512 x 16B units
    for (int u = tid; u < 512; u += 128) {
      int row = u >> 4, c8 = (u & 15) * 8;
      H8 v;
      v.u = *(const uint4*)(qkv + (size_t)(kk + row) * QKV_LD + 2 * DMODEL +
                            h * HDIM + c8);
#pragma unroll
      for (int e = 0; e < 8; ++e) Vts[c8 + e][row] = v.h[e];
    }

    if (wave == 0) __builtin_amdgcn_s_wait_tensorcnt(0);
    __syncthreads();

    if (kk + 32 < seg_end)  // global_prefetch_b8 on next V chunk
      __builtin_prefetch(
          qkv + (size_t)(kk + 32 + lane) * QKV_LD + 2 * DMODEL + h * HDIM, 0, 1);

    // ---- S tiles: n-tile 0 = keys [0,16), n-tile 1 = keys [16,32)
    v8f s0 = {}, s1 = {};
#pragma unroll
    for (int dc = 0; dc < 4; ++dc) {
      F16x16 kb0, kb1;
      kb0.u[0] = *(const uint4*)&Ks[lid][dc * 32 + 8 * hi];
      kb0.u[1] = *(const uint4*)&Ks[lid][dc * 32 + 16 + 8 * hi];
      kb1.u[0] = *(const uint4*)&Ks[16 + lid][dc * 32 + 8 * hi];
      kb1.u[1] = *(const uint4*)&Ks[16 + lid][dc * 32 + 16 + 8 * hi];
      s0 = __builtin_amdgcn_wmma_f32_16x16x32_f16(false, qf[dc].v, false, kb0.v,
                                                  (short)0, s0, false, false);
      s1 = __builtin_amdgcn_wmma_f32_16x16x32_f16(false, qf[dc].v, false, kb1.v,
                                                  (short)0, s1, false, false);
    }

    // ---- online softmax; row m=j+8*hi, 16 cols live across the 16-lane group
#pragma unroll
    for (int j = 0; j < 8; ++j) {
      float t0 = s0[j], t1 = s1[j];
      float mx = fmaxf(t0, t1);
#pragma unroll
      for (int msk = 8; msk >= 1; msk >>= 1)
        mx = fmaxf(mx, __shfl_xor(mx, msk, 32));
      float m_new = fmaxf(m_run[j], mx);
      float alpha = __expf(m_run[j] - m_new);
      float p0 = __expf(t0 - m_new), p1 = __expf(t1 - m_new);
      float rs = p0 + p1;
#pragma unroll
      for (int msk = 8; msk >= 1; msk >>= 1) rs += __shfl_xor(rs, msk, 32);
      l_run[j] = l_run[j] * alpha + rs;
      m_run[j] = m_new;
#pragma unroll
      for (int c = 0; c < 8; ++c) O[c][j] *= alpha;
      Ps[wave][j + 8 * hi][lid]      = (_Float16)p0;
      Ps[wave][j + 8 * hi][16 + lid] = (_Float16)p1;
    }
    // wave-private LDS tile: wait for our DS stores before cross-lane reads
    asm volatile("s_wait_dscnt 0" ::: "memory");

    // rebuild P as an A-layout fragment
    F16x16 pf;
    pf.u[0] = *(const uint4*)&Ps[wave][lid][8 * hi];
    pf.u[1] = *(const uint4*)&Ps[wave][lid][16 + 8 * hi];

    // ---- O(16x128) += P(16x32) * V(32x128)
#pragma unroll
    for (int c = 0; c < 8; ++c) {
      F16x16 vb;
      vb.u[0] = *(const uint4*)&Vts[16 * c + lid][8 * hi];
      vb.u[1] = *(const uint4*)&Vts[16 * c + lid][16 + 8 * hi];
      O[c] = __builtin_amdgcn_wmma_f32_16x16x32_f16(false, pf.v, false, vb.v,
                                                    (short)0, O[c], false, false);
    }
    __syncthreads();  // K/V tiles reused next chunk
  }

  // normalize and store f16 attention output (row-major T x D, head-interleaved)
#pragma unroll
  for (int j = 0; j < 8; ++j) {
    float inv = 1.0f / l_run[j];
    int r = qt0 + j + 8 * hi;
#pragma unroll
    for (int c = 0; c < 8; ++c)
      out[(size_t)r * DMODEL + h * HDIM + 16 * c + lid] =
          (_Float16)(O[c][j] * inv);
  }
}

// ---------------------------------------------------------------------------
// Host-side orchestration (all on `stream`, workspace carved from d_ws)
// ---------------------------------------------------------------------------
extern "C" void kernel_launch(void* const* d_in, const int* in_sizes, int n_in,
                              void* d_out, int out_size, void* d_ws, size_t ws_size,
                              hipStream_t stream) {
  (void)in_sizes; (void)n_in; (void)out_size; (void)ws_size;
  const float* hidden = (const float*)d_in[0];
  const int*   cu     = (const int*)d_in[1];
  const float* rpe    = (const float*)d_in[2];
  const float* wqkv   = (const float*)d_in[3];
  const float* wproj  = (const float*)d_in[4];
  float* out = (float*)d_out;

  char* ws = (char*)d_ws;
  size_t off = 0;
  auto carve = [&](size_t bytes) {
    void* p = ws + off;
    off += (bytes + 255) & ~(size_t)255;
    return p;
  };
  _Float16* hidden_h = (_Float16*)carve((size_t)T_TOK * DMODEL * 2);      // 12.6 MB
  _Float16* wqkv_h   = (_Float16*)carve((size_t)QKV_LD * DMODEL * 2);     // 14.2 MB
  _Float16* wproj_h  = (_Float16*)carve((size_t)DMODEL * DMODEL * 2);     //  4.7 MB
  _Float16* qkv_h    = (_Float16*)carve((size_t)T_TOK * QKV_LD * 2);      // 37.7 MB
  _Float16* attn_h   = (_Float16*)carve((size_t)T_TOK * DMODEL * 2);      // 12.6 MB

  // 1) f32 -> f16 conversions
  {
    int n1 = T_TOK * DMODEL, n2 = QKV_LD * DMODEL, n3 = DMODEL * DMODEL;
    cvt_f32_to_f16<<<(n1 + 255) / 256, 256, 0, stream>>>(hidden, hidden_h, n1);
    cvt_f32_to_f16<<<(n2 + 255) / 256, 256, 0, stream>>>(wqkv, wqkv_h, n2);
    cvt_f32_to_f16<<<(n3 + 255) / 256, 256, 0, stream>>>(wproj, wproj_h, n3);
  }

  // 2) QKV projection: (4096x1536) @ (4608x1536)^T -> f16 (4096x4608)
  {
    dim3 grid(T_TOK / 128, QKV_LD / 128);
    gemm_wmma<true><<<grid, 256, 0, stream>>>(hidden_h, wqkv_h, qkv_h, nullptr,
                                              T_TOK, QKV_LD, DMODEL);
  }

  // 3) RoPE in place on q,k (q also picks up HD^-0.5)
  {
    int n = T_TOK * NH * 64;
    rope_kernel<<<(n + 255) / 256, 256, 0, stream>>>(qkv_h, rpe);
  }

  // 4) flash attention per (64-row tile, head)
  {
    dim3 grid(T_TOK / 64, NH);
    attn_kernel<<<grid, 128, 0, stream>>>(qkv_h, cu, attn_h);
  }

  // 5) output projection: (4096x1536) @ (1536x1536)^T -> f32 d_out
  {
    dim3 grid(T_TOK / 128, DMODEL / 128);
    gemm_wmma<false><<<grid, 256, 0, stream>>>(attn_h, wproj_h, nullptr, out,
                                               T_TOK, DMODEL, DMODEL);
  }
}
